// BahdanauAttn_47072841564502
// MI455X (gfx1250) — matrix operand order repaired and checked by reading, not compiled, not run on hardware
//
#include <hip/hip_runtime.h>
#include <cstddef>

#define HD 1024
#define TT 2048
#define BB 32
#define LDSP 1032      // padded LDS row pitch (elements) -> avoids bank conflicts
#define WG   512       // 16 waves of 32

typedef __attribute__((ext_vector_type(8)))  __bf16 v8bf;
typedef __attribute__((ext_vector_type(16))) __bf16 v16bf;
typedef __attribute__((ext_vector_type(8)))  float  v8f;
typedef __attribute__((ext_vector_type(4)))  float  v4f;   // native vec4 for NT loads

__device__ __forceinline__ v16bf join16(v8bf a, v8bf b) {
  return __builtin_shufflevector(a, b, 0,1,2,3,4,5,6,7,8,9,10,11,12,13,14,15);
}

__device__ __forceinline__ float fast_tanh(float x) {
#if __has_builtin(__builtin_amdgcn_tanhf)
  return __builtin_amdgcn_tanhf(x);
#elif __has_builtin(__builtin_amdgcn_tanh_f32)
  return __builtin_amdgcn_tanh_f32(x);
#else
  return tanhf(x);
#endif
}

// ---------------------------------------------------------------------------
// Pre-pass 1: split W_e (fp32, HxH row-major [g][h]) into bf16 hi/lo planes.
// ---------------------------------------------------------------------------
__global__ __launch_bounds__(256) void prep_we_split(const float* __restrict__ We,
                                                     unsigned short* __restrict__ WeHu,
                                                     unsigned short* __restrict__ WeLu) {
  __bf16* WeH = (__bf16*)WeHu;
  __bf16* WeL = (__bf16*)WeLu;
  int idx = blockIdx.x * 256 + threadIdx.x;          // HD*HD/4 threads
  v4f w = ((const v4f*)We)[idx];
  int base = idx * 4;
#pragma unroll
  for (int i = 0; i < 4; ++i) {
    float x = w[i];
    __bf16 h = (__bf16)x;
    WeH[base + i] = h;
    WeL[base + i] = (__bf16)(x - (float)h);
  }
}

// ---------------------------------------------------------------------------
// Pre-pass 2: hpe[b][g] = sum_h hidden[b][h]*W_h[g][h] + b_h[g] + b_e[g]
// Exact fp32 (tiny: 67 MFLOP, W_h stays in L2).
// ---------------------------------------------------------------------------
__global__ __launch_bounds__(256) void prep_hpe(const float* __restrict__ hidden,
                                                const float* __restrict__ Wh,
                                                const float* __restrict__ bh,
                                                const float* __restrict__ be,
                                                float* __restrict__ hpe) {
  int idx = blockIdx.x * 256 + threadIdx.x;          // BB*HD threads
  int b = idx >> 10;
  int g = idx & (HD - 1);
  const v4f* hr = (const v4f*)(hidden + (size_t)b * HD);
  const v4f* wr = (const v4f*)(Wh + (size_t)g * HD);
  float acc = 0.f;
#pragma unroll 4
  for (int i = 0; i < HD / 4; ++i) {
    v4f hv = hr[i], wv = wr[i];
    acc += hv[0] * wv[0] + hv[1] * wv[1] + hv[2] * wv[2] + hv[3] * wv[3];
  }
  hpe[idx] = acc + bh[g] + be[g];
}

// ---------------------------------------------------------------------------
// Main fused kernel: one workgroup per t.
//   A tile  = encoder[t]  (32 rows x 1024 K) -> LDS as bf16 hi/lo (NT loads:
//             encoder is a one-shot 268 MB stream; keep L2 for W_e reuse)
//   GEMM    : D[b][g] = sum_h A[b][h] * W_e[g][h]   via split bf16 WMMA
//   Epilog  : scores[b][t] = sum_g v[g] * tanh(D + hpe[b][g])
// ---------------------------------------------------------------------------
__global__ __launch_bounds__(WG) void attn_main(const float* __restrict__ enc,
                                                const unsigned short* __restrict__ WeHu,
                                                const unsigned short* __restrict__ WeLu,
                                                const float* __restrict__ hpe,
                                                const float* __restrict__ vvec,
                                                float* __restrict__ scores) {
  extern __shared__ __bf16 smem[];
  __bf16* sAh = smem;                                   // BB*LDSP
  __bf16* sAl = smem + BB * LDSP;                       // BB*LDSP
  float*  sP  = (float*)(smem + 2 * BB * LDSP);         // [16 waves][32 rows]

  const __bf16* WeH = (const __bf16*)WeHu;
  const __bf16* WeL = (const __bf16*)WeLu;

  const int t    = blockIdx.x;
  const int tid  = threadIdx.x;
  const int lane = tid & 31;
  const int wave = tid >> 5;

  // ---- cooperative load + hi/lo split of A tile (32x1024 fp32 = 128 KB) ----
  const v4f* src = (const v4f*)(enc + (size_t)t * (BB * HD));
#pragma unroll
  for (int j = 0; j < 16; ++j) {
    int v4  = tid + j * WG;            // 0..8191
    v4f d = __builtin_nontemporal_load(&src[v4]);
    int row = v4 >> 8;                 // 256 vec4 per row
    int col = (v4 & 255) << 2;
    __bf16* ph = sAh + row * LDSP + col;
    __bf16* pl = sAl + row * LDSP + col;
#pragma unroll
    for (int i = 0; i < 4; ++i) {
      float x = d[i];
      __bf16 h = (__bf16)x;
      ph[i] = h;
      pl[i] = (__bf16)(x - (float)h);
    }
  }
  __syncthreads();

  const int nbase = wave * 64;         // each wave owns 64 output columns
  const int nlane = lane & 15;
  const int khalf = lane >> 4;         // 0: lanes 0-15, 1: lanes 16-31

  v8f acc[2][4];
#pragma unroll
  for (int mt = 0; mt < 2; ++mt)
#pragma unroll
    for (int nt = 0; nt < 4; ++nt) {
      v8f z = {};
      acc[mt][nt] = z;
    }

  // ---- K loop: 3-term split-bf16 WMMA accumulation ----
  for (int k0 = 0; k0 < HD; k0 += 32) {
    // A frags (16-bit A 16x32 layout: per-lane chunks K{0..7,16..23}/{8..15,24..31})
    const int ka = k0 + (khalf << 3);
    v16bf aH[2], aL[2];
#pragma unroll
    for (int mt = 0; mt < 2; ++mt) {
      const __bf16* bh_ = sAh + (mt * 16 + nlane) * LDSP + ka;
      const __bf16* bl_ = sAl + (mt * 16 + nlane) * LDSP + ka;
      aH[mt] = join16(*(const v8bf*)bh_, *(const v8bf*)(bh_ + 16));
      aL[mt] = join16(*(const v8bf*)bl_, *(const v8bf*)(bl_ + 16));
    }
    // B frags (same K order as A: per-lane chunks K{0..7,16..23}/{8..15,24..31})
    const int kb = k0 + (khalf << 3);
#pragma unroll
    for (int nt = 0; nt < 4; ++nt) {
      int g = nbase + nt * 16 + nlane;           // B col n == W_e row g
      const __bf16* pbh = WeH + (size_t)g * HD + kb;
      const __bf16* pbl = WeL + (size_t)g * HD + kb;
      v16bf bH = join16(*(const v8bf*)pbh, *(const v8bf*)(pbh + 16));
      v16bf bL = join16(*(const v8bf*)pbl, *(const v8bf*)(pbl + 16));
#pragma unroll
      for (int mt = 0; mt < 2; ++mt) {
        acc[mt][nt] = __builtin_amdgcn_wmma_f32_16x16x32_bf16(
            false, aH[mt], false, bH, (short)0, acc[mt][nt], false, false);
        acc[mt][nt] = __builtin_amdgcn_wmma_f32_16x16x32_bf16(
            false, aH[mt], false, bL, (short)0, acc[mt][nt], false, false);
        acc[mt][nt] = __builtin_amdgcn_wmma_f32_16x16x32_bf16(
            false, aL[mt], false, bH, (short)0, acc[mt][nt], false, false);
      }
    }
  }

  // ---- fused epilogue: tanh(acc + hpe) dotted with v, reduce over g ----
  float p[2][8];
#pragma unroll
  for (int mt = 0; mt < 2; ++mt)
#pragma unroll
    for (int i = 0; i < 8; ++i) p[mt][i] = 0.f;

#pragma unroll
  for (int nt = 0; nt < 4; ++nt) {
    int g = nbase + nt * 16 + nlane;
    float vg = vvec[g];
#pragma unroll
    for (int mt = 0; mt < 2; ++mt) {
#pragma unroll
      for (int i = 0; i < 8; ++i) {
        int m = mt * 16 + (khalf << 3) + i;      // m == batch b
        float e = fast_tanh(acc[mt][nt][i] + hpe[(size_t)m * HD + g]);
        p[mt][i] += vg * e;
      }
    }
  }

  // reduce over the 16 lanes of each half (n dimension); halves hold distinct rows
#pragma unroll
  for (int mt = 0; mt < 2; ++mt)
#pragma unroll
    for (int i = 0; i < 8; ++i) {
      float s = p[mt][i];
      s += __shfl_xor(s, 8, 32);
      s += __shfl_xor(s, 4, 32);
      s += __shfl_xor(s, 2, 32);
      s += __shfl_xor(s, 1, 32);
      p[mt][i] = s;
    }
  if (nlane == 0) {                                   // lanes 0 and 16
#pragma unroll
    for (int mt = 0; mt < 2; ++mt)
#pragma unroll
      for (int i = 0; i < 8; ++i)
        sP[wave * BB + mt * 16 + (khalf << 3) + i] = p[mt][i];
  }
  __syncthreads();

  // deterministic cross-wave reduction (fixed order, no float atomics)
  if (tid < BB) {
    float s = 0.f;
#pragma unroll
    for (int w = 0; w < 16; ++w) s += sP[w * BB + tid];
    scores[(size_t)tid * TT + t] = s;                 // scores[b][t]
  }
}

// ---------------------------------------------------------------------------
// Softmax over T per batch: out[b][t] = softmax_t(scores[b][t])
// ---------------------------------------------------------------------------
__global__ __launch_bounds__(256) void softmax_k(const float* __restrict__ scores,
                                                 float* __restrict__ out) {
  int b = blockIdx.x, tid = threadIdx.x;
  __shared__ float red[256];
  const float* s = scores + (size_t)b * TT;
  float* o = out + (size_t)b * TT;

  float m = -3.402823e38f;
  for (int t = tid; t < TT; t += 256) m = fmaxf(m, s[t]);
  red[tid] = m;
  __syncthreads();
  for (int off = 128; off > 0; off >>= 1) {
    if (tid < off) red[tid] = fmaxf(red[tid], red[tid + off]);
    __syncthreads();
  }
  m = red[0];
  __syncthreads();

  float sum = 0.f;
  for (int t = tid; t < TT; t += 256) {
    float e = __expf(s[t] - m);
    o[t] = e;
    sum += e;
  }
  red[tid] = sum;
  __syncthreads();
  for (int off = 128; off > 0; off >>= 1) {
    if (tid < off) red[tid] += red[tid + off];
    __syncthreads();
  }
  float inv = 1.0f / red[0];
  for (int t = tid; t < TT; t += 256) o[t] *= inv;
}

// ---------------------------------------------------------------------------
extern "C" void kernel_launch(void* const* d_in, const int* in_sizes, int n_in,
                              void* d_out, int out_size, void* d_ws, size_t ws_size,
                              hipStream_t stream) {
  (void)in_sizes; (void)n_in; (void)out_size; (void)ws_size;
  const float* hidden = (const float*)d_in[0];   // (1,B,H)
  const float* enc    = (const float*)d_in[1];   // (T,B,H)
  const float* Wh     = (const float*)d_in[2];   // (H,H)
  const float* bh     = (const float*)d_in[3];   // (H,)
  const float* We     = (const float*)d_in[4];   // (H,H)
  const float* be     = (const float*)d_in[5];   // (H,)
  const float* vv     = (const float*)d_in[6];   // (H,)
  float* out = (float*)d_out;                    // (B,1,T)

  char* ws = (char*)d_ws;
  unsigned short* WeH = (unsigned short*)ws;                                  // 2 MB
  unsigned short* WeL = (unsigned short*)(ws + (size_t)HD * HD * 2);          // 2 MB
  float* hpe = (float*)(ws + (size_t)HD * HD * 4);                            // 128 KB
  float* sc  = (float*)(ws + (size_t)HD * HD * 4 + (size_t)BB * HD * 4);      // 256 KB

  prep_we_split<<<(HD * HD / 4) / 256, 256, 0, stream>>>(We, WeH, WeL);
  prep_hpe<<<(BB * HD) / 256, 256, 0, stream>>>(hidden, Wh, bh, be, hpe);

  size_t lds_bytes = (size_t)2 * BB * LDSP * 2 + (size_t)16 * BB * 4;  // 134,144 B
  attn_main<<<TT, WG, lds_bytes, stream>>>(enc, WeH, WeL, hpe, vv, sc);

  softmax_k<<<BB, 256, 0, stream>>>(sc, out);
}
